// RBFKernel_65481071402720
// MI455X (gfx1250) — matrix operand order repaired and checked by reading, not compiled, not run on hardware
//
#include <hip/hip_runtime.h>
#include <hip/hip_bf16.h>

// CDNA5 (gfx1250) wave32 WMMA types
typedef __attribute__((ext_vector_type(16))) __bf16 v16bf;
typedef __attribute__((ext_vector_type(8)))  __bf16 v8bf;
typedef __attribute__((ext_vector_type(4)))  __bf16 v4bf;
typedef __attribute__((ext_vector_type(8)))  float  v8f;

#define RBF_D 128   // feature dimension (fixed by the reference)

// ---------------------------------------------------------------------------
// Pre-pass: one wave32 per row of a (rows x 128) f32 matrix.
// Each lane owns 4 contiguous floats (float4 load) and
//   - writes bf16 hi = bf16(v) and lo = bf16(v - hi)  (split-precision GEMM
//     operands, converted ONCE here instead of per-workgroup in the GEMM)
//   - accumulates the row sum-of-squares in f32, butterfly-reduced across
//     the wave; lane 0 writes the row norm.
// ---------------------------------------------------------------------------
__global__ __launch_bounds__(256) void rbf_cvt_rowsq(const float* __restrict__ v,
                                                     __bf16* __restrict__ hi,
                                                     __bf16* __restrict__ lo,
                                                     float*  __restrict__ sq,
                                                     int nrows) {
    const int wavesPerBlock = blockDim.x >> 5;
    const int row  = blockIdx.x * wavesPerBlock + (threadIdx.x >> 5);
    const int lane = threadIdx.x & 31;
    if (row >= nrows) return;

    const float4* p = reinterpret_cast<const float4*>(v + (size_t)row * RBF_D);
    float4 a = p[lane];                       // 32 lanes * 4 floats = 128
    float f[4] = {a.x, a.y, a.z, a.w};

    v4bf h, l;
    float s = 0.0f;
    #pragma unroll
    for (int i = 0; i < 4; ++i) {
        float  vi = f[i];
        __bf16 hb = (__bf16)vi;
        h[i] = hb;
        l[i] = (__bf16)(vi - (float)hb);
        s += vi * vi;
    }
    const size_t off = (size_t)row * RBF_D + lane * 4;
    *reinterpret_cast<v4bf*>(hi + off) = h;   // 8-byte stores
    *reinterpret_cast<v4bf*>(lo + off) = l;

    #pragma unroll
    for (int o = 16; o > 0; o >>= 1)
        s += __shfl_xor(s, o, 32);
    if (lane == 0) sq[row] = s;
}

// Stitch two 8-element bf16 loads into a WMMA v16bf operand (register-level,
// no conversion VALU).
__device__ __forceinline__ v16bf cat16(v8bf a, v8bf b) {
    return __builtin_shufflevector(a, b, 0, 1, 2, 3, 4, 5, 6, 7,
                                         8, 9, 10, 11, 12, 13, 14, 15);
}

// ---------------------------------------------------------------------------
// Main RBF kernel (bf16 operands pre-converted, f32 accumulation).
//   Block: 256 threads = 8 waves, arranged 4 (M) x 2 (N).
//   Wave : 2x2 tiles of 16x16 -> 32x32 output patch; block covers 128x64.
//   K loop: 32-wide chunks, 3 bf16 WMMAs per tile pair (hi*hi + hi*lo + lo*hi).
// Operand layouts per the CDNA5 ISA (CDNA5 ISA), h = (lane>=16):
//   A (16x32 bf16): lane row M=lane&15; 8 bf16 at k0+8h, 8 bf16 at k0+16+8h
//   B (32x16 bf16): lane col N=lane&15; 8 bf16 at k0+8h, 8 bf16 at k0+16+8h
//   C/D (16x16 f32): VGPR r, lane L -> row r+8h, col L&15
// ---------------------------------------------------------------------------
__global__ __launch_bounds__(256) void rbf_wmma(const __bf16* __restrict__ xh,
                                                const __bf16* __restrict__ xl,
                                                const __bf16* __restrict__ yh,
                                                const __bf16* __restrict__ yl,
                                                const float* __restrict__ xsq,
                                                const float* __restrict__ ysq,
                                                const float* __restrict__ gammap,
                                                float* __restrict__ out,
                                                int N, int M) {
    const int lane   = threadIdx.x & 31;
    const int wave   = threadIdx.x >> 5;
    const int lane15 = lane & 15;
    const int h      = lane >> 4;        // half-wave select

    const int wm = wave >> 1;            // 0..3
    const int wn = wave & 1;             // 0..1
    const int m0 = blockIdx.y * 128 + wm * 32;
    const int n0 = blockIdx.x * 64  + wn * 32;

    const float gamma = gammap[0];

    v8f acc[2][2];
    const v8f vzero = {};
    #pragma unroll
    for (int i = 0; i < 2; ++i)
        #pragma unroll
        for (int j = 0; j < 2; ++j)
            acc[i][j] = vzero;

    // Per-lane base offsets (elements) into the bf16 row-major arrays
    const size_t aoff0 = (size_t)(m0 + 0 * 16 + lane15) * RBF_D + 8 * h;
    const size_t aoff1 = (size_t)(m0 + 1 * 16 + lane15) * RBF_D + 8 * h;
    const size_t boff0 = (size_t)(n0 + 0 * 16 + lane15) * RBF_D + 8 * h;
    const size_t boff1 = (size_t)(n0 + 1 * 16 + lane15) * RBF_D + 8 * h;

    #pragma unroll
    for (int k0 = 0; k0 < RBF_D; k0 += 32) {
        v16bf a_hi[2], a_lo[2], b_hi[2], b_lo[2];

        // A operands: two 16-byte loads each (k0+8h and k0+16+8h)
        {
            const v8bf* p;
            p = reinterpret_cast<const v8bf*>(xh + aoff0 + k0);
            a_hi[0] = cat16(p[0], p[2]);            // +0 and +16 elements
            p = reinterpret_cast<const v8bf*>(xl + aoff0 + k0);
            a_lo[0] = cat16(p[0], p[2]);
            p = reinterpret_cast<const v8bf*>(xh + aoff1 + k0);
            a_hi[1] = cat16(p[0], p[2]);
            p = reinterpret_cast<const v8bf*>(xl + aoff1 + k0);
            a_lo[1] = cat16(p[0], p[2]);
        }
        // B operands: two 16-byte loads each (k0+8h and k0+16+8h)
        {
            const v8bf* p;
            p = reinterpret_cast<const v8bf*>(yh + boff0 + k0);
            b_hi[0] = cat16(p[0], p[2]);
            p = reinterpret_cast<const v8bf*>(yl + boff0 + k0);
            b_lo[0] = cat16(p[0], p[2]);
            p = reinterpret_cast<const v8bf*>(yh + boff1 + k0);
            b_hi[1] = cat16(p[0], p[2]);
            p = reinterpret_cast<const v8bf*>(yl + boff1 + k0);
            b_lo[1] = cat16(p[0], p[2]);
        }

        // 3-term split-precision accumulation: hi*hi + hi*lo + lo*hi
        #pragma unroll
        for (int i = 0; i < 2; ++i)
            #pragma unroll
            for (int j = 0; j < 2; ++j) {
                acc[i][j] = __builtin_amdgcn_wmma_f32_16x16x32_bf16(
                    false, a_hi[i], false, b_hi[j], (short)0, acc[i][j], false, false);
                acc[i][j] = __builtin_amdgcn_wmma_f32_16x16x32_bf16(
                    false, a_hi[i], false, b_lo[j], (short)0, acc[i][j], false, false);
                acc[i][j] = __builtin_amdgcn_wmma_f32_16x16x32_bf16(
                    false, a_lo[i], false, b_hi[j], (short)0, acc[i][j], false, false);
            }
    }

    // Epilogue: sq = x2 + y2 - 2*cross, clamp, exp(-gamma*sq); stream stores
    // with non-temporal hint so the 256 MB output doesn't evict the 8 MB of
    // L2-resident bf16 operands.
    #pragma unroll
    for (int i = 0; i < 2; ++i) {
        float xs[8];
        #pragma unroll
        for (int r = 0; r < 8; ++r)
            xs[r] = xsq[m0 + i * 16 + 8 * h + r];

        #pragma unroll
        for (int j = 0; j < 2; ++j) {
            const int n  = n0 + j * 16 + lane15;
            const float ys = ysq[n];
            #pragma unroll
            for (int r = 0; r < 8; ++r) {
                float sq = xs[r] + ys - 2.0f * acc[i][j][r];
                sq = fmaxf(sq, 0.0f);
                const int m = m0 + i * 16 + 8 * h + r;
                __builtin_nontemporal_store(__expf(-gamma * sq),
                                            out + (size_t)m * M + n);
            }
        }
    }
}

extern "C" void kernel_launch(void* const* d_in, const int* in_sizes, int n_in,
                              void* d_out, int out_size, void* d_ws, size_t ws_size,
                              hipStream_t stream) {
    const float* x      = (const float*)d_in[0];
    const float* y      = (const float*)d_in[1];
    const float* gammap = (const float*)d_in[2];
    float* out = (float*)d_out;

    const int N = in_sizes[0] / RBF_D;   // 8192 rows of x
    const int M = in_sizes[1] / RBF_D;   // 8192 rows of y

    // Workspace layout (bf16 operand arrays + f32 row norms): ~8.06 MB
    char* ws = (char*)d_ws;
    __bf16* xh = (__bf16*)ws;                         ws += (size_t)N * RBF_D * 2;
    __bf16* xl = (__bf16*)ws;                         ws += (size_t)N * RBF_D * 2;
    __bf16* yh = (__bf16*)ws;                         ws += (size_t)M * RBF_D * 2;
    __bf16* yl = (__bf16*)ws;                         ws += (size_t)M * RBF_D * 2;
    float*  xsq = (float*)ws;                         ws += (size_t)N * 4;
    float*  ysq = (float*)ws;

    // Pre-pass: bf16 hi/lo conversion + row norms (8 waves per block)
    rbf_cvt_rowsq<<<(N + 7) / 8, 256, 0, stream>>>(x, xh, xl, xsq, N);
    rbf_cvt_rowsq<<<(M + 7) / 8, 256, 0, stream>>>(y, yh, yl, ysq, M);

    // Main GEMM+exp: block covers 128 (rows of x) x 64 (rows of y)
    dim3 grid(M / 64, N / 128);
    rbf_wmma<<<grid, 256, 0, stream>>>(xh, xl, yh, yl, xsq, ysq, gammap, out, N, M);
}
